// Hawk_74835510166241
// MI455X (gfx1250) — matrix operand order repaired and checked by reading, not compiled, not run on hardware
//
#include <hip/hip_runtime.h>

typedef __attribute__((ext_vector_type(16))) __bf16 v16bf;
typedef __attribute__((ext_vector_type(8)))  __bf16 v8bf;
typedef __attribute__((ext_vector_type(8)))  float  v8f;
typedef __attribute__((ext_vector_type(4)))  unsigned int u32x4;
typedef __attribute__((ext_vector_type(8)))  int i32x8;
typedef __attribute__((ext_vector_type(4)))  int i32x4;

constexpr int Bd = 32768;   // batch
constexpr int Hd = 1024;    // hidden
constexpr int MT = 128;     // block tile M
constexpr int NT = 64;      // block tile N
constexpr int KT = 32;      // K chunk (one bf16 WMMA depth)
constexpr int LSTR = 40;    // LDS row stride in elements (64B data + 16B pad = 80B)
constexpr float EPSC = 1e-4f;

// ---------------- TDM: 2D tile load, global -> LDS (clang-23 6-arg builtin) --------------
// D# per CDNA5 ISA ch.8: group0 = flags/lds/global addr, group1 = dims/strides/pad.
// cols is always 32 bf16 (64B = 16 DWORDs) here: pad_interval code 3 (16 DW),
// pad_amount code 3 (4 DW = 16B) -> LDS row pitch 80B, matching LSTR.
static __device__ __forceinline__ void tdm_load_tile(uint32_t lds_off, const void* gptr,
                                                     uint32_t rows, uint32_t cols,
                                                     uint32_t stride_elems) {
  uint64_t ga = (uint64_t)(uintptr_t)gptr;
  u32x4 g0;
  g0[0] = 1u;                                   // count=1, user descriptor
  g0[1] = lds_off;                              // LDS byte address
  g0[2] = (uint32_t)ga;                         // global_addr[31:0]
  g0[3] = (uint32_t)((ga >> 32) & 0x01FFFFFFu)  // global_addr[56:32]
        | 0x80000000u;                          // type=2 ("image")
  i32x8 g1;
  g1[0] = (int)((1u << 16)      // data_size = 2 bytes
              | (1u << 20)      // pad_enable
              | (3u << 22)      // pad_interval: 16 DWORDs (64B)
              | (3u << 25));    // pad_amount: 4 DWORDs (16B)
  g1[1] = (int)((cols & 0xffffu) << 16);                                  // tensor_dim0 lo
  g1[2] = (int)(((cols >> 16) & 0xffffu) | ((rows & 0xffffu) << 16));     // dim0 hi | dim1 lo
  g1[3] = (int)(((rows >> 16) & 0xffffu) | ((cols & 0xffffu) << 16));     // dim1 hi | tile_dim0
  g1[4] = (int)(rows & 0xffffu);                                          // tile_dim1 (tile_dim2=0)
  g1[5] = (int)stride_elems;                                              // tensor_dim0_stride lo32
  g1[6] = 0;
  g1[7] = 0;
  i32x4 z4 = {0, 0, 0, 0};
  i32x8 z8 = {0, 0, 0, 0, 0, 0, 0, 0};
  __builtin_amdgcn_tensor_load_to_lds(g0, g1, z4, z4, z8, 0);
}

// ---------------- WMMA fragment loads from LDS (ISA 7.12.2 layouts) --------------
union FragU { v16bf v; v8bf h[2]; };

// A 16x32 bf16: lane l holds row M=l; group g: VGPR0-3 = K 8g..8g+7, VGPR4-7 = K 16+8g..
static __device__ __forceinline__ v16bf load_a_frag(const __bf16* sm, int row, int g) {
  const __bf16* p = sm + row * LSTR + 8 * g;
  FragU f;
  f.h[0] = *(const v8bf*)p;
  f.h[1] = *(const v8bf*)(p + 16);
  return f.v;
}

// B 32x16 bf16: lane l holds col N=l; group g: VGPR0-3 = K 8g..8g+7, VGPR4-7 = K 16+8g..
static __device__ __forceinline__ v16bf load_b_frag(const __bf16* sm, int row, int g) {
  const __bf16* p = sm + row * LSTR + 8 * g;
  FragU f;
  f.h[0] = *(const v8bf*)p;
  f.h[1] = *(const v8bf*)(p + 16);
  return f.v;
}

static __device__ __forceinline__ v8f wmma_bf16(v16bf a, v16bf b, v8f c) {
  return __builtin_amdgcn_wmma_f32_16x16x32_bf16(false, a, false, b, (short)0, c, false, false);
}

static __device__ __forceinline__ float sigmoidf_(float x) { return 1.0f / (1.0f + expf(-x)); }
static __device__ __forceinline__ float geluf_(float x) {
  return 0.5f * x * (1.0f + erff(x * 0.70710678118654752f));
}

// ---------------- fp32 -> bf16 conversion --------------
__global__ void hawk_cvt_bf16(const float* __restrict__ s, __bf16* __restrict__ d, int n) {
  int i = blockIdx.x * blockDim.x + threadIdx.x;
  int stride = gridDim.x * blockDim.x;
  for (; i < n; i += stride) d[i] = (__bf16)s[i];
}

// ---------------- Kernel 1: gate = gelu(x@gate_w^T); xr = x@rec_w^T; conv + state roll ----
__global__ __launch_bounds__(256) void hawk_k1(
    const __bf16* __restrict__ xbf, const __bf16* __restrict__ wg, const __bf16* __restrict__ wr,
    const float* __restrict__ conv_state, const float* __restrict__ conv_w,
    const float* __restrict__ conv_b, __bf16* __restrict__ gate_bf, __bf16* __restrict__ conv_bf,
    float* __restrict__ ncs) {
  __shared__ alignas(16) __bf16 sA[MT * LSTR];
  __shared__ alignas(16) __bf16 sB0[NT * LSTR];
  __shared__ alignas(16) __bf16 sB1[NT * LSTR];
  const int tid = threadIdx.x;
  const int wave = tid >> 5;
  const int lane = tid & 31;
  const int l = lane & 15, g = lane >> 4;
  const int m0 = blockIdx.y * MT;
  const int n0 = blockIdx.x * NT;
  const int wm = (wave >> 1) * 32;
  const int wn = (wave & 1) * 32;

  const uint32_t ldsA = (uint32_t)(uintptr_t)&sA[0];
  const uint32_t ldsB0 = (uint32_t)(uintptr_t)&sB0[0];
  const uint32_t ldsB1 = (uint32_t)(uintptr_t)&sB1[0];

  v8f zero = {};
  v8f accG[2][2], accR[2][2];
#pragma unroll
  for (int mi = 0; mi < 2; mi++)
#pragma unroll
    for (int ni = 0; ni < 2; ni++) { accG[mi][ni] = zero; accR[mi][ni] = zero; }

  for (int kt = 0; kt < Hd; kt += KT) {
    __syncthreads();
    if (wave == 0) {
      tdm_load_tile(ldsA, xbf + (size_t)m0 * Hd + kt, MT, KT, Hd);
      tdm_load_tile(ldsB0, wg + (size_t)n0 * Hd + kt, NT, KT, Hd);
      tdm_load_tile(ldsB1, wr + (size_t)n0 * Hd + kt, NT, KT, Hd);
      __builtin_amdgcn_s_wait_tensorcnt(0);
    }
    __syncthreads();
    v16bf a[2], b0[2], b1[2];
#pragma unroll
    for (int i = 0; i < 2; i++) {
      a[i] = load_a_frag(sA, wm + i * 16 + l, g);
      b0[i] = load_b_frag(sB0, wn + i * 16 + l, g);
      b1[i] = load_b_frag(sB1, wn + i * 16 + l, g);
    }
#pragma unroll
    for (int mi = 0; mi < 2; mi++)
#pragma unroll
      for (int ni = 0; ni < 2; ni++) {
        accG[mi][ni] = wmma_bf16(a[mi], b0[ni], accG[mi][ni]);
        accR[mi][ni] = wmma_bf16(a[mi], b1[ni], accR[mi][ni]);
      }
  }

  // epilogue: D layout — lane l = col, group g + vgpr r -> row 8g+r
#pragma unroll
  for (int mi = 0; mi < 2; mi++) {
#pragma unroll
    for (int ni = 0; ni < 2; ni++) {
      const int n = n0 + wn + ni * 16 + l;
      const float w0 = conv_w[n * 4 + 0], w1 = conv_w[n * 4 + 1];
      const float w2 = conv_w[n * 4 + 2], w3 = conv_w[n * 4 + 3];
      const float bias = conv_b[n];
      const int mbase = m0 + wm + mi * 16 + 8 * g;
#pragma unroll
      for (int r = 0; r < 8; r++) {
        const int m = mbase + r;
        const size_t idx = (size_t)m * Hd + n;
        const size_t cb = idx * 3;
        const float gv = accG[mi][ni][r];
        const float xr = accR[mi][ni][r];
        const float cs0 = conv_state[cb + 0];
        const float cs1 = conv_state[cb + 1];
        const float cs2 = conv_state[cb + 2];
        const float conv = fmaf(cs0, w0, fmaf(cs1, w1, fmaf(cs2, w2, fmaf(xr, w3, bias))));
        gate_bf[idx] = (__bf16)geluf_(gv);
        conv_bf[idx] = (__bf16)conv;
        ncs[cb + 0] = cs1;
        ncs[cb + 1] = cs2;
        ncs[cb + 2] = xr;
      }
    }
  }
}

// ---------------- Kernel 2: i/r gates + RG-LRU state update; y = gate*h --------------
__global__ __launch_bounds__(256) void hawk_k2(
    const __bf16* __restrict__ cbf, const __bf16* __restrict__ wig, const __bf16* __restrict__ wrg,
    const float* __restrict__ rglru, const float* __restrict__ base_param,
    const __bf16* __restrict__ gate_bf, float* __restrict__ nrs, __bf16* __restrict__ ybf) {
  __shared__ alignas(16) __bf16 sA[MT * LSTR];
  __shared__ alignas(16) __bf16 sB0[NT * LSTR];
  __shared__ alignas(16) __bf16 sB1[NT * LSTR];
  const int tid = threadIdx.x;
  const int wave = tid >> 5;
  const int lane = tid & 31;
  const int l = lane & 15, g = lane >> 4;
  const int m0 = blockIdx.y * MT;
  const int n0 = blockIdx.x * NT;
  const int wm = (wave >> 1) * 32;
  const int wn = (wave & 1) * 32;

  const uint32_t ldsA = (uint32_t)(uintptr_t)&sA[0];
  const uint32_t ldsB0 = (uint32_t)(uintptr_t)&sB0[0];
  const uint32_t ldsB1 = (uint32_t)(uintptr_t)&sB1[0];

  v8f zero = {};
  v8f accI[2][2], accR[2][2];
#pragma unroll
  for (int mi = 0; mi < 2; mi++)
#pragma unroll
    for (int ni = 0; ni < 2; ni++) { accI[mi][ni] = zero; accR[mi][ni] = zero; }

  for (int kt = 0; kt < Hd; kt += KT) {
    __syncthreads();
    if (wave == 0) {
      tdm_load_tile(ldsA, cbf + (size_t)m0 * Hd + kt, MT, KT, Hd);
      tdm_load_tile(ldsB0, wig + (size_t)n0 * Hd + kt, NT, KT, Hd);
      tdm_load_tile(ldsB1, wrg + (size_t)n0 * Hd + kt, NT, KT, Hd);
      __builtin_amdgcn_s_wait_tensorcnt(0);
    }
    __syncthreads();
    v16bf a[2], b0[2], b1[2];
#pragma unroll
    for (int i = 0; i < 2; i++) {
      a[i] = load_a_frag(sA, wm + i * 16 + l, g);
      b0[i] = load_b_frag(sB0, wn + i * 16 + l, g);
      b1[i] = load_b_frag(sB1, wn + i * 16 + l, g);
    }
#pragma unroll
    for (int mi = 0; mi < 2; mi++)
#pragma unroll
      for (int ni = 0; ni < 2; ni++) {
        accI[mi][ni] = wmma_bf16(a[mi], b0[ni], accI[mi][ni]);
        accR[mi][ni] = wmma_bf16(a[mi], b1[ni], accR[mi][ni]);
      }
  }

#pragma unroll
  for (int mi = 0; mi < 2; mi++) {
#pragma unroll
    for (int ni = 0; ni < 2; ni++) {
      const int n = n0 + wn + ni * 16 + l;
      float base = sigmoidf_(base_param[n]);
      base = fminf(fmaxf(base, EPSC), 1.0f - EPSC);
      const float lb = logf(base);
      const int mbase = m0 + wm + mi * 16 + 8 * g;
#pragma unroll
      for (int r = 0; r < 8; r++) {
        const int m = mbase + r;
        const size_t idx = (size_t)m * Hd + n;
        const float it = sigmoidf_(accI[mi][ni][r]);
        const float rt = sigmoidf_(accR[mi][ni][r]);
        const float a_t = expf(8.0f * rt * lb);           // base ** (C*r_t)
        const float mult = sqrtf(fmaxf(1.0f - a_t * a_t, 0.0f));
        const float conv = (float)cbf[idx];
        const float h = rglru[idx] * a_t + mult * (it * conv);
        nrs[idx] = h;
        ybf[idx] = (__bf16)((float)gate_bf[idx] * h);
      }
    }
  }
}

// ---------------- Kernel 3: out = y @ out_w^T --------------
__global__ __launch_bounds__(256) void hawk_k3(const __bf16* __restrict__ ybf,
                                               const __bf16* __restrict__ wout,
                                               float* __restrict__ out) {
  __shared__ alignas(16) __bf16 sA[MT * LSTR];
  __shared__ alignas(16) __bf16 sB0[NT * LSTR];
  const int tid = threadIdx.x;
  const int wave = tid >> 5;
  const int lane = tid & 31;
  const int l = lane & 15, g = lane >> 4;
  const int m0 = blockIdx.y * MT;
  const int n0 = blockIdx.x * NT;
  const int wm = (wave >> 1) * 32;
  const int wn = (wave & 1) * 32;

  const uint32_t ldsA = (uint32_t)(uintptr_t)&sA[0];
  const uint32_t ldsB0 = (uint32_t)(uintptr_t)&sB0[0];

  v8f zero = {};
  v8f acc[2][2];
#pragma unroll
  for (int mi = 0; mi < 2; mi++)
#pragma unroll
    for (int ni = 0; ni < 2; ni++) acc[mi][ni] = zero;

  for (int kt = 0; kt < Hd; kt += KT) {
    __syncthreads();
    if (wave == 0) {
      tdm_load_tile(ldsA, ybf + (size_t)m0 * Hd + kt, MT, KT, Hd);
      tdm_load_tile(ldsB0, wout + (size_t)n0 * Hd + kt, NT, KT, Hd);
      __builtin_amdgcn_s_wait_tensorcnt(0);
    }
    __syncthreads();
    v16bf a[2], b[2];
#pragma unroll
    for (int i = 0; i < 2; i++) {
      a[i] = load_a_frag(sA, wm + i * 16 + l, g);
      b[i] = load_b_frag(sB0, wn + i * 16 + l, g);
    }
#pragma unroll
    for (int mi = 0; mi < 2; mi++)
#pragma unroll
      for (int ni = 0; ni < 2; ni++) acc[mi][ni] = wmma_bf16(a[mi], b[ni], acc[mi][ni]);
  }

#pragma unroll
  for (int mi = 0; mi < 2; mi++) {
#pragma unroll
    for (int ni = 0; ni < 2; ni++) {
      const int n = n0 + wn + ni * 16 + l;
      const int mbase = m0 + wm + mi * 16 + 8 * g;
#pragma unroll
      for (int r = 0; r < 8; r++) {
        out[(size_t)(mbase + r) * Hd + n] = acc[mi][ni][r];
      }
    }
  }
}

extern "C" void kernel_launch(void* const* d_in, const int* in_sizes, int n_in, void* d_out,
                              int out_size, void* d_ws, size_t ws_size, hipStream_t stream) {
  (void)in_sizes; (void)n_in; (void)out_size; (void)ws_size;
  const float* x = (const float*)d_in[0];
  const float* conv_state = (const float*)d_in[1];
  const float* rglru = (const float*)d_in[2];
  const float* gate_w = (const float*)d_in[3];
  const float* rec_w = (const float*)d_in[4];
  const float* conv_w = (const float*)d_in[5];
  const float* conv_b = (const float*)d_in[6];
  const float* ig_w = (const float*)d_in[7];
  const float* rg_w = (const float*)d_in[8];
  const float* base_param = (const float*)d_in[9];
  const float* out_w = (const float*)d_in[10];

  const size_t BH = (size_t)Bd * Hd;   // 32M elements
  const size_t HH = (size_t)Hd * Hd;   // 1M elements
  char* ws = (char*)d_ws;
  __bf16* x_bf = (__bf16*)(ws);
  __bf16* gate_bf = (__bf16*)(ws + 2 * BH);
  __bf16* conv_bf = (__bf16*)(ws + 4 * BH);
  __bf16* y_bf = (__bf16*)(ws + 6 * BH);
  __bf16* wg_bf = (__bf16*)(ws + 8 * BH);
  __bf16* wr_bf = wg_bf + HH;
  __bf16* wig_bf = wr_bf + HH;
  __bf16* wrg_bf = wig_bf + HH;
  __bf16* wout_bf = wrg_bf + HH;

  float* out = (float*)d_out;
  float* ncs = out + BH;          // new_conv_state [B,H,3]
  float* nrs = out + 4 * BH;      // new_rglru_state [B,H]

  hawk_cvt_bf16<<<4096, 256, 0, stream>>>(x, x_bf, (int)BH);
  hawk_cvt_bf16<<<512, 256, 0, stream>>>(gate_w, wg_bf, (int)HH);
  hawk_cvt_bf16<<<512, 256, 0, stream>>>(rec_w, wr_bf, (int)HH);
  hawk_cvt_bf16<<<512, 256, 0, stream>>>(ig_w, wig_bf, (int)HH);
  hawk_cvt_bf16<<<512, 256, 0, stream>>>(rg_w, wrg_bf, (int)HH);
  hawk_cvt_bf16<<<512, 256, 0, stream>>>(out_w, wout_bf, (int)HH);

  dim3 grid(Hd / NT, Bd / MT);  // (16, 256)
  hawk_k1<<<grid, 256, 0, stream>>>(x_bf, wg_bf, wr_bf, conv_state, conv_w, conv_b, gate_bf,
                                    conv_bf, ncs);
  hawk_k2<<<grid, 256, 0, stream>>>(conv_bf, wig_bf, wrg_bf, rglru, base_param, gate_bf, nrs,
                                    y_bf);
  hawk_k3<<<grid, 256, 0, stream>>>(y_bf, wout_bf, out);
}
